// RBFLinear_55911884259445
// MI455X (gfx1250) — matrix operand order repaired and checked by reading, not compiled, not run on hardware
//
#include <hip/hip_runtime.h>
#include <math.h>

typedef __attribute__((ext_vector_type(16))) _Float16 v16h;
typedef __attribute__((ext_vector_type(8)))  float    v8f;

// Problem constants (match reference)
#define NROWS   32768
#define IN_DIM  64
#define HID     2048
#define NTILES  (HID / 16)        // 128 center tiles of 16
#define ROWS_PER_WG 128           // 8 waves * 16 rows
#define THREADS 256

// LDS layout:
//   [0, 256KB):  B fragments, f16, pre-swizzled into WMMA layout:
//                [tile(128)][kmat(2)][lane(32)][16 halfs]  (32 B per lane-fragment)
//   [256KB, 288KB): float4 per center: {c_sq, gamma^2*log2(e), W0, W1}
#define LDS_B_BYTES (NTILES * 2 * 32 * 16 * 2)   // 262144
#define LDS_S_BYTES (HID * 16)                   // 32768
#define LDS_TOTAL   (LDS_B_BYTES + LDS_S_BYTES)  // 294912 < 320KB WGP LDS

#define LOG2E 1.4426950408889634f

__global__ void rbf_linear_fused(const float* __restrict__ x,
                                 const float* __restrict__ centers,
                                 const float* __restrict__ gammas,
                                 const float* __restrict__ W,
                                 const float* __restrict__ bias,
                                 float* __restrict__ out) {
    extern __shared__ char smem[];
    _Float16* ldsB = (_Float16*)smem;
    float4*   ldsS = (float4*)(smem + LDS_B_BYTES);

    const int tid  = threadIdx.x;
    const int lane = tid & 31;
    const int wave = tid >> 5;
    const int half = lane >> 4;       // 0: lanes 0-15, 1: lanes 16-31
    const int lmod = lane & 15;

    // ---------------- Stage 1: cooperative LDS staging (once per WG) -------
    // B fragments: group g = (tile, kmat, lane); each group holds K = 8h+j (j<8),
    // 16+8h+(j-8) (j>=8) of one center column, converted f32->f16.
    for (int g = tid; g < NTILES * 2 * 32; g += THREADS) {
        const int t  = g >> 6;                 // tile
        const int m  = (g >> 5) & 1;           // kmat (K base = 32*m)
        const int gl = g & 31;                 // lane this fragment feeds
        const int c  = t * 16 + (gl & 15);     // center index (B column)
        const int kb = m * 32 + (gl >> 4) * 8;   // K base for this half-wave
        const float* src = centers + (size_t)c * IN_DIM + kb;
        _Float16* dst = ldsB + (size_t)g * 16;
        #pragma unroll
        for (int j = 0; j < 16; ++j) dst[j] = (_Float16)src[j < 8 ? j : j + 8];
    }
    // Per-center scalars: ||c||^2, gamma^2 * log2(e) (folds the exp->exp2
    // prescale into the table so the hot loop is a bare v_exp_f32), W rows.
    for (int c = tid; c < HID; c += THREADS) {
        const float* cc = centers + (size_t)c * IN_DIM;
        float s = 0.0f;
        #pragma unroll 8
        for (int k = 0; k < IN_DIM; ++k) s = fmaf(cc[k], cc[k], s);
        const float g = gammas[c];
        ldsS[c] = make_float4(s, g * g * LOG2E, W[c], W[HID + c]);
    }
    __syncthreads();

    // ---------------- Stage 2: per-wave row tile ---------------------------
    const int r0 = blockIdx.x * ROWS_PER_WG + wave * 16;

    // ||x_r||^2 for this lane's nominal row (r0 + lane%16); broadcast later.
    float xs = 0.0f;
    {
        const float* xq = x + (size_t)(r0 + lmod) * IN_DIM;
        #pragma unroll 8
        for (int k = 0; k < IN_DIM; ++k) xs = fmaf(xq[k], xq[k], xs);
    }
    // Lane's 8 accumulator rows are M = v + 8*half; fetch their ||x||^2.
    float xsq[8];
    #pragma unroll
    for (int v = 0; v < 8; ++v) xsq[v] = __shfl(xs, v + 8 * half, 32);

    // A fragments (16x32 f16, ISA layout): lane holds row M = lane%16.
    // VGPR v in 0..3 -> K = half*8 + 2v (+0,1); v in 4..7 -> K = 16 + half*8 + 2(v-4).
    v16h a0, a1;
    {
        const float* xr = x + (size_t)(r0 + lmod) * IN_DIM;
        #pragma unroll
        for (int v = 0; v < 8; ++v) {
            const int base = (v < 4) ? (half * 8 + 2 * v)
                                     : (16 + half * 8 + 2 * (v - 4));
            a0[2 * v]     = (_Float16)xr[base];
            a0[2 * v + 1] = (_Float16)xr[base + 1];
            a1[2 * v]     = (_Float16)xr[32 + base];
            a1[2 * v + 1] = (_Float16)xr[32 + base + 1];
        }
    }

    float acc0[8] = {0.f, 0.f, 0.f, 0.f, 0.f, 0.f, 0.f, 0.f};
    float acc1[8] = {0.f, 0.f, 0.f, 0.f, 0.f, 0.f, 0.f, 0.f};

    #pragma unroll 2
    for (int t = 0; t < NTILES; ++t) {
        const v16h b0 = *(const v16h*)(ldsB + ((size_t)(t * 2 + 0) * 32 + lane) * 16);
        const v16h b1 = *(const v16h*)(ldsB + ((size_t)(t * 2 + 1) * 32 + lane) * 16);
        const float4 sc = ldsS[t * 16 + lmod];   // {c_sq, g2*log2e, W0, W1} for column N

        v8f c = {};
        // D = A0*B0 + 0 ; D += A1*B1  (K = 0..31, 32..63)
        c = __builtin_amdgcn_wmma_f32_16x16x32_f16(false, a0, false, b0,
                                                   (short)0, c, false, false);
        c = __builtin_amdgcn_wmma_f32_16x16x32_f16(false, a1, false, b1,
                                                   (short)0, c, false, false);

        #pragma unroll
        for (int v = 0; v < 8; ++v) {
            float sq = xsq[v] + sc.x - 2.0f * c[v];
            sq = fmaxf(sq, 0.0f);
            // exp(-sq*g^2) == 2^(-sq * g^2*log2e); raw v_exp_f32, no guard.
            // Argument is always <= 0; subnormal tails flush to 0 (desired).
            const float kern = __builtin_amdgcn_exp2f(-sq * sc.y);
            acc0[v] = fmaf(kern, sc.z, acc0[v]);
            acc1[v] = fmaf(kern, sc.w, acc1[v]);
        }
    }

    // ---------------- Stage 3: reduce over the 16 N-columns (lanes) --------
    #pragma unroll
    for (int v = 0; v < 8; ++v) {
        #pragma unroll
        for (int off = 1; off < 16; off <<= 1) {   // stays within each half-wave
            acc0[v] += __shfl_xor(acc0[v], off, 32);
            acc1[v] += __shfl_xor(acc1[v], off, 32);
        }
    }
    if (lmod == 0) {   // lane 0 -> rows 0..7, lane 16 -> rows 8..15
        const float b0v = bias[0], b1v = bias[1];
        const int rbase = r0 + half * 8;
        #pragma unroll
        for (int v = 0; v < 8; ++v) {
            float2 o = make_float2(acc0[v] + b0v, acc1[v] + b1v);
            *(float2*)(out + (size_t)(rbase + v) * 2) = o;
        }
    }
}

extern "C" void kernel_launch(void* const* d_in, const int* in_sizes, int n_in,
                              void* d_out, int out_size, void* d_ws, size_t ws_size,
                              hipStream_t stream) {
    (void)in_sizes; (void)n_in; (void)d_ws; (void)ws_size; (void)out_size;
    const float* x       = (const float*)d_in[0];
    const float* centers = (const float*)d_in[1];
    const float* gammas  = (const float*)d_in[2];
    const float* W       = (const float*)d_in[3];
    const float* bias    = (const float*)d_in[4];
    float* out = (float*)d_out;

    dim3 grid(NROWS / ROWS_PER_WG);   // 256 workgroups
    dim3 block(THREADS);              // 8 waves (wave32)
    hipLaunchKernelGGL(rbf_linear_fused, grid, block, LDS_TOTAL, stream,
                       x, centers, gammas, W, bias, out);
}
